// InferenceNetwork_90666759618872
// MI455X (gfx1250) — matrix operand order repaired and checked by reading, not compiled, not run on hardware
//
#include <hip/hip_runtime.h>
#include <hip/hip_bf16.h>
#include <math.h>

typedef __bf16 bf16_t;
typedef __attribute__((ext_vector_type(16))) __bf16 v16bf;
typedef __attribute__((ext_vector_type(8)))  __bf16 v8bf;
typedef __attribute__((ext_vector_type(4)))  __bf16 v4bf;
typedef __attribute__((ext_vector_type(8)))  float  v8f;

#define B_    128
#define T_    512
#define H_    1024
#define S_    128
#define KK_   64
#define HID_  512
#define DG_   (H_ + KK_)     // 1088 fused gi K-dim: [x_t | state]
#define G3H_  (3 * H_)       // 3072

#define NBLK  64             // 8 row-groups x 8 column-slices
#define NTHR  256            // 8 waves

// ---------------------------------------------------------------------------
// Grid-wide sense barrier
// ---------------------------------------------------------------------------
__device__ __forceinline__ void gsync(unsigned* cnt, unsigned* gen) {
  __threadfence();
  __syncthreads();
  if (threadIdx.x == 0) {
    unsigned g = __atomic_load_n(gen, __ATOMIC_ACQUIRE);
    if (atomicAdd(cnt, 1u) == (unsigned)(gridDim.x - 1)) {
      __atomic_store_n(cnt, 0u, __ATOMIC_RELAXED);
      __atomic_store_n(gen, g + 1u, __ATOMIC_RELEASE);
    } else {
      while (__atomic_load_n(gen, __ATOMIC_ACQUIRE) == g) {
        __builtin_amdgcn_s_sleep(1);
      }
    }
  }
  __syncthreads();
}

// ---------------------------------------------------------------------------
// Async global->LDS staging (CDNA5 ISA ch.10 async ops, ASYNCcnt)
// ---------------------------------------------------------------------------
__device__ __forceinline__ void async_ld_b128(void* lds_dst, const void* gbase,
                                              unsigned byte_off) {
  unsigned lds = (unsigned)(uintptr_t)lds_dst;
  unsigned long long sa = (unsigned long long)(uintptr_t)gbase;
  asm volatile("global_load_async_to_lds_b128 %0, %1, %2"
               :: "v"(lds), "v"(byte_off), "s"(sa) : "memory");
}
__device__ __forceinline__ void wait_async() {
  asm volatile("s_wait_asynccnt 0x0" ::: "memory");
}

// ---------------------------------------------------------------------------
// WMMA fragment loaders (ISA 7.12.2 wave32 layouts)
// A 16x32: lane(row=L&15, h=L>>4): e0..7 -> K=h*8+e ; e8..15 -> K=16+h*8+e
// B 32x16: lane(col=L&15, h=L>>4): e0..7 -> K=h*8+e ; e8..15 -> K=16+h*8+e
// C tile : row = v + 8*(L>>4), col = L&15
// ---------------------------------------------------------------------------
__device__ __forceinline__ v16bf frag_rows(const bf16_t* __restrict__ base,
                                           int ld, int r0, int k0, int lane) {
  const bf16_t* p = base + (size_t)(r0 + (lane & 15)) * ld + k0 + (lane >> 4) * 8;
  v8bf lo = *(const v8bf*)(p);
  v8bf hi = *(const v8bf*)(p + 16);
  v16bf f;
#pragma unroll
  for (int i = 0; i < 8; ++i) { f[i] = lo[i]; f[i + 8] = hi[i]; }
  return f;
}
__device__ __forceinline__ v16bf frag_cols(const bf16_t* __restrict__ base,
                                           int ld, int n0, int k0, int lane) {
  const bf16_t* p = base + (size_t)(n0 + (lane & 15)) * ld + k0 + (lane >> 4) * 8;
  v8bf lo = *(const v8bf*)(p);
  v8bf hi = *(const v8bf*)(p + 16);
  v16bf f;
#pragma unroll
  for (int i = 0; i < 8; ++i) { f[i] = lo[i]; f[i + 8] = hi[i]; }
  return f;
}
__device__ __forceinline__ v8f wmma_bf16(v16bf a, v16bf b, v8f c) {
  return __builtin_amdgcn_wmma_f32_16x16x32_bf16(false, a, false, b,
                                                 (short)0, c, false, false);
}
__device__ __forceinline__ float sigmoidf_(float x) {
  return 1.0f / (1.0f + __expf(-x));
}

// ---------------------------------------------------------------------------
struct Params {
  const float* contexts;   // [B,T,H] f32
  const float* b_ih;       // [3H]
  const float* b_hh;       // [3H]
  const float* b1;         // [HID]
  const float* b2;         // [K]
  const bf16_t* wWg;       // [3H, DG]  fused [W_ih_x | (E.W_ih_emb^T)^T] bf16
  const bf16_t* wWhh;      // [3H, H]   bf16
  const bf16_t* wW1;       // [HID, H]  bf16
  const bf16_t* wW2;       // [K, HID]  bf16
  bf16_t* hbf;             // [2][B,H]  bf16 ping-pong
  bf16_t* a1;              // [B,HID]   bf16
  bf16_t* statebf;         // [B,K]     bf16
  float*  hf;              // [B,H]     f32
  float*  out;             // [B,T,K]   f32
  unsigned* cnt;
  unsigned* gen;
};

__global__ void __launch_bounds__(NTHR, 1) rnn_persistent(Params p) {
  __shared__ __align__(16) bf16_t atile[16 * DG_];  // [x_t | state] rows of mt
  __shared__ __align__(16) bf16_t htile[16 * H_];   // h rows of mt

  const int tid  = threadIdx.x;
  const int lane = tid & 31;
  const int wv   = tid >> 5;           // 0..7
  const int mt   = blockIdx.x >> 3;    // row-group (m-tile) 0..7
  const int sub  = blockIdx.x & 7;     // column slice 0..7
  const int cl   = lane & 15;
  const int hh   = lane >> 4;

  for (int t = 0; t < T_; ++t) {
    const int cur = t & 1, prv = cur ^ 1;
    bf16_t* hb_r = p.hbf + (size_t)prv * B_ * H_;   // h_{t-1}
    bf16_t* hb_w = p.hbf + (size_t)cur * B_ * H_;   // h_t

    // ==== Stage 1: stage h_{t-1}, state, x_t into LDS =====================
    // h rows (raw bf16) : async copy, 2048 x b128
#pragma unroll
    for (int it = 0; it < 8; ++it) {
      int f = tid + it * NTHR;                       // 0..2047
      int r = f >> 7, q = f & 127;
      async_ld_b128(&htile[r * H_ + q * 8], hb_r,
                    (unsigned)(((mt * 16 + r) * H_ + q * 8) * 2));
    }
    // state rows -> atile cols 1024..1087 : async copy, 128 x b128
    if (tid < 128) {
      int r = tid >> 3, q = tid & 7;
      async_ld_b128(&atile[r * DG_ + H_ + q * 8], p.statebf,
                    (unsigned)(((mt * 16 + r) * KK_ + q * 8) * 2));
    }
    // x_t rows : f32 -> bf16 convert through registers (float4 per thread)
    {
      const float4* ctx4 = (const float4*)p.contexts;
      for (int idx = tid; idx < 16 * (H_ / 4); idx += NTHR) {
        int r = idx >> 8, c = (idx & 255) * 4;
        float4 f4 = ctx4[((size_t)(mt * 16 + r) * T_ + t) * (H_ / 4) + (c >> 2)];
        v4bf b4 = { (bf16_t)f4.x, (bf16_t)f4.y, (bf16_t)f4.z, (bf16_t)f4.w };
        *(v4bf*)&atile[r * DG_ + c] = b4;
      }
    }
    wait_async();
    __syncthreads();

    // ==== Stage 2: fused gate GEMM + GRU nonlinearity =====================
    // wave job: 16 h-columns jc*16..+15 -> 6 accumulators (r/z/n x gi/gh)
    {
      const int jc = sub * 8 + wv;        // 0..63
      const int n0 = jc * 16;
      v8f ar = {}, az = {}, an = {}, br = {}, bz = {}, bn = {};
      // gh part: A = htile (h_{t-1}), B = W_hh rows {n0, H+n0, 2H+n0}
#pragma unroll 2
      for (int k0 = 0; k0 < H_; k0 += 32) {
        if ((k0 & 127) == 0) {
          __builtin_prefetch(p.wWhh + (size_t)n0 * H_ + k0 + 256, 0, 1);
          __builtin_prefetch(p.wWhh + (size_t)(H_ + n0) * H_ + k0 + 256, 0, 1);
          __builtin_prefetch(p.wWhh + (size_t)(2 * H_ + n0) * H_ + k0 + 256, 0, 1);
        }
        v16bf fa = frag_rows(htile, H_, 0, k0, lane);
        br = wmma_bf16(fa, frag_cols(p.wWhh, H_, n0,          k0, lane), br);
        bz = wmma_bf16(fa, frag_cols(p.wWhh, H_, H_ + n0,     k0, lane), bz);
        bn = wmma_bf16(fa, frag_cols(p.wWhh, H_, 2 * H_ + n0, k0, lane), bn);
      }
      // gi part: A = atile ([x_t|state]), B = Wg rows {n0, H+n0, 2H+n0}
#pragma unroll 2
      for (int k0 = 0; k0 < DG_; k0 += 32) {
        if ((k0 & 127) == 0) {
          __builtin_prefetch(p.wWg + (size_t)n0 * DG_ + k0 + 256, 0, 1);
          __builtin_prefetch(p.wWg + (size_t)(H_ + n0) * DG_ + k0 + 256, 0, 1);
          __builtin_prefetch(p.wWg + (size_t)(2 * H_ + n0) * DG_ + k0 + 256, 0, 1);
        }
        v16bf fa = frag_rows(atile, DG_, 0, k0, lane);
        ar = wmma_bf16(fa, frag_cols(p.wWg, DG_, n0,          k0, lane), ar);
        az = wmma_bf16(fa, frag_cols(p.wWg, DG_, H_ + n0,     k0, lane), az);
        an = wmma_bf16(fa, frag_cols(p.wWg, DG_, 2 * H_ + n0, k0, lane), an);
      }
      // gate math in registers -> h_new
      const int col = n0 + cl;
      const float bir = p.b_ih[col], biz = p.b_ih[H_ + col], bin = p.b_ih[2 * H_ + col];
      const float bhr = p.b_hh[col], bhz = p.b_hh[H_ + col], bhn = p.b_hh[2 * H_ + col];
#pragma unroll
      for (int v = 0; v < 8; ++v) {
        const int row = mt * 16 + v + 8 * hh;
        const float r = sigmoidf_((ar[v] + bir) + (br[v] + bhr));
        const float z = sigmoidf_((az[v] + biz) + (bz[v] + bhz));
        const float n = tanhf((an[v] + bin) + r * (bn[v] + bhn));
        const float ho = p.hf[(size_t)row * H_ + col];
        const float hn = (1.f - z) * n + z * ho;
        p.hf[(size_t)row * H_ + col] = hn;
        hb_w[(size_t)row * H_ + col] = (bf16_t)hn;
      }
    }
    gsync(p.cnt, p.gen);

    // ==== Stage 3: restage h_t -> LDS, then W1 (tanh head layer) ==========
#pragma unroll
    for (int it = 0; it < 8; ++it) {
      int f = tid + it * NTHR;
      int r = f >> 7, q = f & 127;
      async_ld_b128(&htile[r * H_ + q * 8], hb_w,
                    (unsigned)(((mt * 16 + r) * H_ + q * 8) * 2));
    }
    wait_async();
    __syncthreads();
    if (wv < 4) {
      const int n0 = (sub * 4 + wv) * 16;   // 32 n-tiles across the group
      v8f acc = {};
#pragma unroll 2
      for (int k0 = 0; k0 < H_; k0 += 32) {
        v16bf fa = frag_rows(htile, H_, 0, k0, lane);
        acc = wmma_bf16(fa, frag_cols(p.wW1, H_, n0, k0, lane), acc);
      }
      const float bv = p.b1[n0 + cl];
#pragma unroll
      for (int v = 0; v < 8; ++v) {
        const int row = mt * 16 + v + 8 * hh;
        p.a1[(size_t)row * HID_ + n0 + cl] = (bf16_t)tanhf(acc[v] + bv);
      }
    }
    gsync(p.cnt, p.gen);

    // ==== Stage 4: fused W2 + softmax (one wave per row-group) ============
    if (sub == 0 && wv == 0) {
      v8f acc0 = {}, acc1 = {}, acc2 = {}, acc3 = {};
#pragma unroll 2
      for (int k0 = 0; k0 < HID_; k0 += 32) {
        v16bf fa = frag_rows(p.a1, HID_, mt * 16, k0, lane);
        acc0 = wmma_bf16(fa, frag_cols(p.wW2, HID_, 0,  k0, lane), acc0);
        acc1 = wmma_bf16(fa, frag_cols(p.wW2, HID_, 16, k0, lane), acc1);
        acc2 = wmma_bf16(fa, frag_cols(p.wW2, HID_, 32, k0, lane), acc2);
        acc3 = wmma_bf16(fa, frag_cols(p.wW2, HID_, 48, k0, lane), acc3);
      }
      const float b20 = p.b2[cl],      b21 = p.b2[16 + cl];
      const float b22 = p.b2[32 + cl], b23 = p.b2[48 + cl];
#pragma unroll
      for (int v = 0; v < 8; ++v) {
        const int row = mt * 16 + v + 8 * hh;
        float x0 = acc0[v] + b20, x1 = acc1[v] + b21;
        float x2 = acc2[v] + b22, x3 = acc3[v] + b23;
        float* orow = p.out + ((size_t)row * T_ + t) * KK_;
        orow[cl] = x0; orow[16 + cl] = x1; orow[32 + cl] = x2; orow[48 + cl] = x3;
        // softmax(clip(.)) within the 16-lane half (each half owns 8 rows)
        x0 = fminf(fmaxf(x0, -10.f), 10.f);
        x1 = fminf(fmaxf(x1, -10.f), 10.f);
        x2 = fminf(fmaxf(x2, -10.f), 10.f);
        x3 = fminf(fmaxf(x3, -10.f), 10.f);
        float m = fmaxf(fmaxf(x0, x1), fmaxf(x2, x3));
#pragma unroll
        for (int o = 1; o < 16; o <<= 1) m = fmaxf(m, __shfl_xor(m, o, 32));
        float e0 = __expf(x0 - m), e1 = __expf(x1 - m);
        float e2 = __expf(x2 - m), e3 = __expf(x3 - m);
        float s = (e0 + e1) + (e2 + e3);
#pragma unroll
        for (int o = 1; o < 16; o <<= 1) s += __shfl_xor(s, o, 32);
        const float inv = 1.f / s;
        bf16_t* srow = p.statebf + (size_t)row * KK_;
        srow[cl] = (bf16_t)(e0 * inv); srow[16 + cl] = (bf16_t)(e1 * inv);
        srow[32 + cl] = (bf16_t)(e2 * inv); srow[48 + cl] = (bf16_t)(e3 * inv);
      }
    }
    gsync(p.cnt, p.gen);
  }
}

// ---------------------------------------------------------------------------
// Setup kernels
// ---------------------------------------------------------------------------
__global__ void k_build_wg(const float* __restrict__ W_ih,
                           const float* __restrict__ E,
                           bf16_t* __restrict__ wg, int n) {
  int i = blockIdx.x * blockDim.x + threadIdx.x;
  if (i >= n) return;                       // n = 3H * DG
  int g = i / DG_, c = i % DG_;
  float v;
  if (c < H_) {
    v = W_ih[(size_t)g * (H_ + S_) + c];
  } else {                                  // fold emb: M[k,g] = sum_s E[k,s]*Wih[g,H+s]
    int k = c - H_;
    float acc = 0.f;
    for (int s = 0; s < S_; ++s)
      acc += E[k * S_ + s] * W_ih[(size_t)g * (H_ + S_) + H_ + s];
    v = acc;
  }
  wg[i] = (bf16_t)v;
}
__global__ void k_cvt_bf16(const float* __restrict__ s, bf16_t* __restrict__ d,
                           int n) {
  int i = blockIdx.x * blockDim.x + threadIdx.x;
  if (i < n) d[i] = (bf16_t)s[i];
}
__global__ void k_init_h(float* hf, bf16_t* hbf, int n) {
  int i = blockIdx.x * blockDim.x + threadIdx.x;
  if (i < n) { hf[i] = 0.f; hbf[i] = (bf16_t)0.f; hbf[i + n] = (bf16_t)0.f; }
}
__global__ void k_init_state(const float* __restrict__ s, bf16_t* __restrict__ d,
                             int n) {
  int i = blockIdx.x * blockDim.x + threadIdx.x;
  if (i < n) d[i] = (bf16_t)s[i];
}
__global__ void k_zero_u32(unsigned* p, int n) {
  int i = blockIdx.x * blockDim.x + threadIdx.x;
  if (i < n) p[i] = 0u;
}

// ---------------------------------------------------------------------------
extern "C" void kernel_launch(void* const* d_in, const int* in_sizes, int n_in,
                              void* d_out, int out_size, void* d_ws,
                              size_t ws_size, hipStream_t stream) {
  (void)in_sizes; (void)n_in; (void)out_size; (void)ws_size;
  const float* contexts      = (const float*)d_in[0];
  const float* initial_state = (const float*)d_in[1];
  const float* E             = (const float*)d_in[2];
  const float* W_ih          = (const float*)d_in[3];
  const float* W_hh          = (const float*)d_in[4];
  const float* b_ih          = (const float*)d_in[5];
  const float* b_hh          = (const float*)d_in[6];
  const float* W1            = (const float*)d_in[7];
  const float* b1            = (const float*)d_in[8];
  const float* W2            = (const float*)d_in[9];
  const float* b2            = (const float*)d_in[10];
  float* out = (float*)d_out;

  char* ws = (char*)d_ws;
  size_t off = 0;
  auto alloc = [&](size_t bytes) -> void* {
    off = (off + 255) & ~(size_t)255;
    void* q = ws + off;
    off += bytes;
    return q;
  };
  bf16_t* wWg     = (bf16_t*)alloc((size_t)G3H_ * DG_ * 2);
  bf16_t* wWhh    = (bf16_t*)alloc((size_t)G3H_ * H_ * 2);
  bf16_t* wW1     = (bf16_t*)alloc((size_t)HID_ * H_ * 2);
  bf16_t* wW2     = (bf16_t*)alloc((size_t)KK_ * HID_ * 2);
  bf16_t* hbf     = (bf16_t*)alloc((size_t)2 * B_ * H_ * 2);
  bf16_t* a1      = (bf16_t*)alloc((size_t)B_ * HID_ * 2);
  bf16_t* statebf = (bf16_t*)alloc((size_t)B_ * KK_ * 2);
  float*  hf      = (float*)alloc((size_t)B_ * H_ * 4);
  unsigned* barrier = (unsigned*)alloc(256);

  const int thr = 256;
  auto blocks = [&](int n) { return (n + thr - 1) / thr; };

  k_zero_u32<<<1, 64, 0, stream>>>(barrier, 64);
  k_build_wg<<<blocks(G3H_ * DG_), thr, 0, stream>>>(W_ih, E, wWg, G3H_ * DG_);
  k_cvt_bf16<<<blocks(G3H_ * H_), thr, 0, stream>>>(W_hh, wWhh, G3H_ * H_);
  k_cvt_bf16<<<blocks(HID_ * H_), thr, 0, stream>>>(W1, wW1, HID_ * H_);
  k_cvt_bf16<<<blocks(KK_ * HID_), thr, 0, stream>>>(W2, wW2, KK_ * HID_);
  k_init_h<<<blocks(B_ * H_), thr, 0, stream>>>(hf, hbf, B_ * H_);
  k_init_state<<<blocks(B_ * KK_), thr, 0, stream>>>(initial_state, statebf,
                                                     B_ * KK_);

  Params p;
  p.contexts = contexts;
  p.b_ih = b_ih; p.b_hh = b_hh; p.b1 = b1; p.b2 = b2;
  p.wWg = wWg; p.wWhh = wWhh; p.wW1 = wW1; p.wW2 = wW2;
  p.hbf = hbf; p.a1 = a1; p.statebf = statebf; p.hf = hf;
  p.out = out;
  p.cnt = barrier; p.gen = barrier + 64;

  rnn_persistent<<<NBLK, NTHR, 0, stream>>>(p);
}
